// GCN_l_73203422593432
// MI455X (gfx1250) — matrix operand order repaired and checked by reading, not compiled, not run on hardware
//
#include <hip/hip_runtime.h>
#include <hip/hip_fp16.h>

// GCN forward: 3x (GEMM -> symmetric-norm aggregate -> ReLU) -> GEMM -> softmax
// N=100000 (divisible by 16), E=3200000, F_IN=64, H=128.
// Aggregation accumulator (51MB) is L2-resident on MI455X (192MB L2);
// edge scatter uses global_atomic_add_f32 which resolves at L2.

typedef __attribute__((ext_vector_type(16))) _Float16 v16h;
typedef __attribute__((ext_vector_type(8)))  _Float16 v8h;
typedef __attribute__((ext_vector_type(8)))  float    v8f;

#define NNODES 100000

// ---------------- degree / normalization ----------------
__global__ void k_init_deg(float* __restrict__ deg, int n) {
  int i = blockIdx.x * blockDim.x + threadIdx.x;
  if (i < n) deg[i] = 1.0f;                       // self-loop counts 1
}

__global__ void k_deg_accum(const int* __restrict__ dst, float* __restrict__ deg, int e) {
  int i = blockIdx.x * blockDim.x + threadIdx.x;
  if (i < e) atomicAdd(&deg[dst[i]], 1.0f);
}

__global__ void k_dinv(float* __restrict__ deg, int n) {
  int i = blockIdx.x * blockDim.x + threadIdx.x;
  if (i < n) deg[i] = rsqrtf(deg[i]);             // deg >= 1 always
}

// ---------------- WMMA tiled GEMM: Y[N,HOUT] = act(X)[N,HIN] @ W[HIN,HOUT] ----------------
// One block per 16-row M tile; one wave per 16-col N tile. f16 inputs, f32 accumulate.
// Weights are stored TRANSPOSED in LDS (sWt[n][k]) so every B fragment is two
// contiguous, 16B-aligned ds_load_b128s (mirrors the A fragment path).
template<int HIN, int HOUT, bool RELU, bool BIAS>
__global__ __launch_bounds__(32 * (HOUT / 16))
void k_gemm(const float* __restrict__ X, const float* __restrict__ W,
            const float* __restrict__ bias, float* __restrict__ Y) {
  constexpr int T = 32 * (HOUT / 16);
  __shared__ _Float16 sA [16 * HIN];      // A tile  [m][k], converted (ReLU fused)
  __shared__ _Float16 sWt[HOUT * HIN];    // weights [n][k], converted + transposed

  const int tid   = threadIdx.x;
  const int mbase = blockIdx.x * 16;

  for (int i = tid; i < 16 * HIN; i += T) {
    float v = X[(size_t)(mbase + i / HIN) * HIN + (i % HIN)];
    if (RELU) v = fmaxf(v, 0.0f);
    sA[i] = (_Float16)v;
  }
  for (int i = tid; i < HIN * HOUT; i += T) {
    const int k = i / HOUT, nn = i % HOUT;
    sWt[nn * HIN + k] = (_Float16)W[i];
  }
  __syncthreads();

  const int lane  = tid & 31;
  const int nbase = (tid >> 5) * 16;
  const int m     = lane & 15;            // row within tile (A) / column (B,D)
  const int hi    = lane >> 4;            // lane half selects K-subrange
  const int ncol  = nbase + m;

  v8f acc = {};
  #pragma unroll
  for (int ks = 0; ks < HIN / 32; ++ks) {
    const int kb0 = ks * 32;
    // A 16x32 f16: lane half hi -> K = kb0+hi*8+[0..7] and kb0+16+hi*8+[0..7]
    const v8h a0 = *(const v8h*)&sA[m * HIN + kb0 + hi * 8];
    const v8h a1 = *(const v8h*)&sA[m * HIN + kb0 + 16 + hi * 8];
    const v16h a = __builtin_shufflevector(a0, a1, 0,1,2,3,4,5,6,7,8,9,10,11,12,13,14,15);
    // B 32x16 f16: lane = N column; K = kb0+hi*8+[0..7] and kb0+16+hi*8+[0..7] in sWt
    const v8h b0 = *(const v8h*)&sWt[ncol * HIN + kb0 + hi * 8];
    const v8h b1 = *(const v8h*)&sWt[ncol * HIN + kb0 + 16 + hi * 8];
    const v16h b = __builtin_shufflevector(b0, b1, 0,1,2,3,4,5,6,7,8,9,10,11,12,13,14,15);

    acc = __builtin_amdgcn_wmma_f32_16x16x32_f16(
        /*neg_a=*/false, a, /*neg_b=*/false, b,
        /*c_mod=*/(short)0, acc, /*reuse_a=*/false, /*reuse_b=*/false);
  }

  // D 16x16 f32: VGPR r -> M=r (lanes 0-15), M=r+8 (lanes 16-31); N = lane&15
  #pragma unroll
  for (int r = 0; r < 8; ++r) {
    float v = acc[r];
    if constexpr (BIAS) v += bias[ncol];
    Y[(size_t)(mbase + r + hi * 8) * HOUT + ncol] = v;
  }
}

// ---------------- self-loop + bias init: out = b + h * dinv^2 ----------------
template<int F>
__global__ void k_selfinit(const float* __restrict__ H, const float* __restrict__ dinv,
                           const float* __restrict__ b, float* __restrict__ out, int n) {
  int i = blockIdx.x * blockDim.x + threadIdx.x;        // over n*(F/4)
  if (i >= n * (F / 4)) return;
  const int node = i / (F / 4);
  const int f4   = (i % (F / 4)) * 4;
  float w = dinv[node]; w *= w;
  const float4 h = *(const float4*)(H + (size_t)node * F + f4);
  float4 o;
  o.x = b[f4 + 0] + h.x * w;
  o.y = b[f4 + 1] + h.y * w;
  o.z = b[f4 + 2] + h.z * w;
  o.w = b[f4 + 3] + h.w * w;
  *(float4*)(out + (size_t)node * F + f4) = o;
}

// ---------------- edge scatter: out[dst] += h[src] * dinv[src]*dinv[dst] ----------------
// 32 lanes per edge (4 features/lane for F=128); f32 atomics resolve in L2.
template<int F>
__global__ void k_scatter(const float* __restrict__ H, const int* __restrict__ src,
                          const int* __restrict__ dst, const float* __restrict__ dinv,
                          float* __restrict__ out, int e) {
  constexpr int TPE = F / 4;
  const int g  = blockIdx.x * blockDim.x + threadIdx.x;
  const int ed = g / TPE;
  if (ed >= e) return;
  const int f4 = (g % TPE) * 4;

  // prefetch a future edge's source row to hide random-gather latency
  const int ed2 = ed + 64;
  if (ed2 < e) {
    const int s2 = __ldg(&src[ed2]);
    __builtin_prefetch(H + (size_t)s2 * F + f4, 0, 0);   // global_prefetch_b8
  }

  const int s = src[ed], d = dst[ed];
  const float w = dinv[s] * dinv[d];
  const float4 h = *(const float4*)(H + (size_t)s * F + f4);
  float* o = out + (size_t)d * F + f4;
  atomicAdd(o + 0, h.x * w);
  atomicAdd(o + 1, h.y * w);
  atomicAdd(o + 2, h.z * w);
  atomicAdd(o + 3, h.w * w);
}

// ---------------- row softmax over 64 features, one wave32 per row ----------------
__global__ void k_softmax64(const float* __restrict__ L, float* __restrict__ out, int n) {
  const int row = blockIdx.x * (blockDim.x >> 5) + (threadIdx.x >> 5);
  if (row >= n) return;
  const int lane = threadIdx.x & 31;
  float v0 = L[(size_t)row * 64 + lane];
  float v1 = L[(size_t)row * 64 + 32 + lane];
  float m = fmaxf(v0, v1);
  #pragma unroll
  for (int o = 16; o > 0; o >>= 1) m = fmaxf(m, __shfl_xor(m, o, 32));
  const float e0 = __expf(v0 - m), e1 = __expf(v1 - m);
  float s = e0 + e1;
  #pragma unroll
  for (int o = 16; o > 0; o >>= 1) s += __shfl_xor(s, o, 32);
  const float inv = 1.0f / s;
  out[(size_t)row * 64 + lane]      = e0 * inv;
  out[(size_t)row * 64 + 32 + lane] = e1 * inv;
}

// ---------------- driver ----------------
extern "C" void kernel_launch(void* const* d_in, const int* in_sizes, int n_in,
                              void* d_out, int out_size, void* d_ws, size_t ws_size,
                              hipStream_t stream) {
  (void)n_in; (void)out_size; (void)ws_size;
  const float* x    = (const float*)d_in[0];
  const int*   ei   = (const int*)d_in[1];
  const float* W1   = (const float*)d_in[2];
  const float* b1   = (const float*)d_in[3];
  const float* W2   = (const float*)d_in[4];
  const float* b2   = (const float*)d_in[5];
  const float* W3   = (const float*)d_in[6];
  const float* b3   = (const float*)d_in[7];
  const float* Wout = (const float*)d_in[8];
  const float* bout = (const float*)d_in[9];

  const int n = NNODES;                    // = in_sizes[0] / 64
  const int e = in_sizes[1] / 2;
  const int* src = ei;
  const int* dst = ei + e;

  char*  ws   = (char*)d_ws;
  float* dinv = (float*)ws;                                                  // N floats
  float* HG   = (float*)(ws + (1u << 20));                                   // N*128
  float* AG   = (float*)(ws + (1u << 20) + (size_t)NNODES * 128 * 4);        // N*128
  float* outp = (float*)d_out;                                               // N*64

  const int nb  = (n + 255) / 256;
  const int eb  = (e + 255) / 256;
  const int mb  = n / 16;                  // 6250 exact
  const int ib  = (n * 32 + 255) / 256;    // selfinit threads: n * (128/4)
  const int sb  = (e * 32 + 255) / 256;    // scatter threads:  e * 32
  const int smb = (n + 7) / 8;             // softmax: 8 rows/block

  k_init_deg <<<nb, 256, 0, stream>>>(dinv, n);
  k_deg_accum<<<eb, 256, 0, stream>>>(dst, dinv, e);
  k_dinv     <<<nb, 256, 0, stream>>>(dinv, n);

  // layer 1: h = x@W1 ; a1 = b1 + selfloop + scatter   (ReLU deferred to next GEMM)
  k_gemm<64, 128, false, false><<<mb, 256, 0, stream>>>(x, W1, nullptr, HG);
  k_selfinit<128><<<ib, 256, 0, stream>>>(HG, dinv, b1, AG, n);
  k_scatter <128><<<sb, 256, 0, stream>>>(HG, src, dst, dinv, AG, e);

  // layer 2
  k_gemm<128, 128, true, false><<<mb, 256, 0, stream>>>(AG, W2, nullptr, HG);
  k_selfinit<128><<<ib, 256, 0, stream>>>(HG, dinv, b2, AG, n);
  k_scatter <128><<<sb, 256, 0, stream>>>(HG, src, dst, dinv, AG, e);

  // layer 3
  k_gemm<128, 128, true, false><<<mb, 256, 0, stream>>>(AG, W3, nullptr, HG);
  k_selfinit<128><<<ib, 256, 0, stream>>>(HG, dinv, b3, AG, n);
  k_scatter <128><<<sb, 256, 0, stream>>>(HG, src, dst, dinv, AG, e);

  // output layer: logits = relu(a3)@Wout + bout ; softmax rows
  k_gemm<128, 64, true, true><<<mb, 128, 0, stream>>>(AG, Wout, bout, HG);
  k_softmax64<<<smb, 256, 0, stream>>>(HG, outp, n);
}
